// SimpleBiLSTM_22110491640587
// MI455X (gfx1250) — matrix operand order repaired and checked by reading, not compiled, not run on hardware
//
#include <hip/hip_runtime.h>

typedef __bf16 bf16;
typedef bf16  v16bf __attribute__((ext_vector_type(16)));
typedef bf16  v8bf  __attribute__((ext_vector_type(8)));
typedef float v8f   __attribute__((ext_vector_type(8)));
typedef unsigned int u32x4 __attribute__((ext_vector_type(4)));
typedef int i32x4 __attribute__((ext_vector_type(4)));
typedef int i32x8 __attribute__((ext_vector_type(8)));

#define B_   32
#define T_   512
#define D_   512
#define H_   512
#define G4_  2048        // 4*H
#define NT_  8           // 16-wide n-tiles per 128-col slice
#define KS_  16          // 32-wide k-steps over K=512
#define NBLK_PER_DIR 16  // hidden slices of 32
#define TPB  256

union V16 { v16bf v; v8bf h[2]; };

// ---------------------------------------------------------------------------
// Tensor Data Mover: DMA a contiguous run of `nelem` 2-byte elements from
// global memory into LDS. 1-D descriptor: tensor_dim0 = tile_dim0 = nelem,
// tile_dim1 = 1. Issue from ONE wave; that wave must s_wait_tensorcnt(0).
// ---------------------------------------------------------------------------
__device__ __forceinline__ void tdm_load_b16_1d(const bf16* gsrc, bf16* ldst,
                                                int nelem) {
  const unsigned lds_off = (unsigned)(unsigned long long)(void*)ldst;
  const unsigned long long ga = (unsigned long long)(const void*)gsrc;
  u32x4 g0 = {};
  g0[0] = 1u;                                            // count=1, user mode
  g0[1] = lds_off;                                       // lds_addr
  g0[2] = (unsigned)ga;                                  // global_addr[31:0]
  g0[3] = (unsigned)((ga >> 32) & 0x01ffffffu) | (2u << 30); // addr[56:32]|type=2
  i32x8 g1 = {};
  g1[0] = 0x00010000;                                    // data_size=1 (2B)
  g1[1] = (nelem & 0xffff) << 16;                        // tensor_dim0[15:0]
  g1[2] = ((nelem >> 16) & 0xffff) | (1 << 16);          // dim0[31:16] | dim1=1
  g1[3] = (nelem & 0xffff) << 16;                        // tile_dim0 = nelem
  g1[4] = 1;                                             // tile_dim1=1, tile_dim2=0
  g1[5] = nelem;                                         // dim0_stride[31:0]
  g1[6] = 0;
  g1[7] = 0;
  i32x4 z4 = {};
#if defined(__clang_major__) && (__clang_major__ >= 23)
  i32x8 z8 = {};
  __builtin_amdgcn_tensor_load_to_lds(g0, g1, z4, z4, z8, 0);
#else
  __builtin_amdgcn_tensor_load_to_lds(g0, g1, z4, z4, 0);
#endif
}

// ---------------------------------------------------------------------------
// Swizzled preload of a 512 x 128 (f32 -> bf16) weight slice into LDS in
// WMMA B-fragment layout: sB[((ks*NT + nt)*32 + lane)*16 + j]
// ---------------------------------------------------------------------------
__device__ __forceinline__ void load_B_slice(const float* __restrict__ W, int ldn,
                                             int4 nb, bf16* __restrict__ sB) {
  for (int e = threadIdx.x; e < KS_ * NT_ * 32 * 16; e += blockDim.x) {
    const int lane = e & 31;
    const int j    = (e >> 5) & 15;
    const int nt   = (e >> 9) & (NT_ - 1);
    const int ks   = e >> 12;
    const int k    = ks * 32 + ((lane >> 4) << 4) + j;
    const int nc   = nt * 16 + (lane & 15);          // local column 0..127
    const int ch   = nc >> 5;
    const int n    = ((ch == 0) ? nb.x : (ch == 1) ? nb.y : (ch == 2) ? nb.z : nb.w)
                     + (nc & 31);
    sB[((ks * NT_ + nt) * 32 + lane) * 16 + j] = (bf16)W[(size_t)k * ldn + n];
  }
}

// ---------------------------------------------------------------------------
// Core tile: acc[32 x 128] += A[32 x 512](LDS, row-major bf16) * B_slice(LDS)
// 8 waves; wave w owns n-tile w (16 cols), both m-tiles. 32 WMMAs per wave.
// ---------------------------------------------------------------------------
__device__ __forceinline__ void compute_tile(const bf16* __restrict__ sA,
                                             const bf16* __restrict__ sB,
                                             v8f& acc0, v8f& acc1) {
  const int lane = threadIdx.x & 31;
  const int wave = threadIdx.x >> 5;
  const int half = lane >> 4;
  const int mrow = lane & 15;
  #pragma unroll 4
  for (int ks = 0; ks < KS_; ++ks) {
    const v16bf bfrag = *(const v16bf*)(sB + ((ks * NT_ + wave) * 32 + lane) * 16);
    V16 a0, a1;
    const bf16* pa0 = sA + mrow * D_ + ks * 32 + half * 16;         // K: half*16..+7
    a0.h[0] = *(const v8bf*)(pa0);                                  // VGPR 0..3
    a0.h[1] = *(const v8bf*)(pa0 + 8);                              // K: half*16+8..
    const bf16* pa1 = sA + (16 + mrow) * D_ + ks * 32 + half * 16;
    a1.h[0] = *(const v8bf*)(pa1);
    a1.h[1] = *(const v8bf*)(pa1 + 8);
    acc0 = __builtin_amdgcn_wmma_f32_16x16x32_bf16(false, a0.v, false, bfrag,
                                                   (short)0, acc0, false, false);
    acc1 = __builtin_amdgcn_wmma_f32_16x16x32_bf16(false, a1.v, false, bfrag,
                                                   (short)0, acc1, false, false);
  }
}

// ---------------------------------------------------------------------------
// Prep: x -> bf16, flipped x -> bf16 (src(t) = (len+T-1-t) mod T),
// zero h double-buffers and the device-barrier counters (every launch).
// ---------------------------------------------------------------------------
__global__ void prep_kernel(const float* __restrict__ x, const int* __restrict__ lengths,
                            bf16* __restrict__ xb, bf16* __restrict__ xrb,
                            bf16* __restrict__ hbuf, unsigned* __restrict__ bar) {
  const size_t i = (size_t)blockIdx.x * TPB + threadIdx.x;
  const size_t NTOT = (size_t)B_ * T_ * D_;
  if (i < NTOT) {
    const int d = (int)(i % D_);
    const int t = (int)((i / D_) % T_);
    const int b = (int)(i / ((size_t)T_ * D_));
    xb[i] = (bf16)x[i];
    const int L  = lengths[b];
    const int ts = (L + T_ - 1 - t) & (T_ - 1);
    xrb[i] = (bf16)x[((size_t)b * T_ + ts) * D_ + d];
  }
  if (i < (size_t)2 * 2 * B_ * H_) hbuf[i] = (bf16)0.0f;
  if (i < 64) bar[i] = 0u;
}

// ---------------------------------------------------------------------------
// Phase 1: xg = x @ Wx + b  (both directions), WMMA bf16 -> f32.
// grid = (16 n-slices, 64 m-chunks of 256 rows, 2 dirs)
// ---------------------------------------------------------------------------
__global__ void __launch_bounds__(TPB)
proj_gemm_kernel(const bf16* __restrict__ xb, const bf16* __restrict__ xrb,
                 const float* __restrict__ Wx_f, const float* __restrict__ Wx_b,
                 const float* __restrict__ b_f, const float* __restrict__ b_b,
                 float* __restrict__ xg_f, float* __restrict__ xg_b) {
  __shared__ bf16 sB[KS_ * NT_ * 32 * 16];   // 128 KB swizzled weight slice
  __shared__ bf16 sA[32 * D_];               // 32 KB activation sub-tile
  const int nslice = blockIdx.x;
  const int mblk   = blockIdx.y;
  const int dir    = blockIdx.z;
  const bf16*  X  = dir ? xrb : xb;
  const float* W  = dir ? Wx_b : Wx_f;
  const float* bv = dir ? b_b  : b_f;
  float*       XG = dir ? xg_b : xg_f;
  const int n0 = nslice * 128;
  load_B_slice(W, G4_, make_int4(n0, n0 + 32, n0 + 64, n0 + 96), sB);
  const int lane = threadIdx.x & 31;
  const int wave = threadIdx.x >> 5;
  const int half = lane >> 4;
  const int n    = n0 + wave * 16 + (lane & 15);
  const float bias = bv[n];
  __syncthreads();
  for (int sub = 0; sub < 8; ++sub) {
    const int m0 = mblk * 256 + sub * 32;
    if (wave == 0) {                         // TDM: one DMA per 32x512 tile
      tdm_load_b16_1d(X + (size_t)m0 * D_, sA, 32 * D_);
      __builtin_amdgcn_s_wait_tensorcnt(0);
    }
    __syncthreads();
    v8f acc0 = {}, acc1 = {};
    compute_tile(sA, sB, acc0, acc1);
    __syncthreads();                 // done with sA before next sub overwrites it
    #pragma unroll
    for (int r = 0; r < 8; ++r) {    // D-frag: row = r + 8*half (+16 for m-tile 1)
      XG[(size_t)(m0 + r + 8 * half) * G4_ + n]      = acc0[r] + bias;
      XG[(size_t)(m0 + 16 + r + 8 * half) * G4_ + n] = acc1[r] + bias;
    }
  }
}

// ---------------------------------------------------------------------------
// Phase 2: persistent recurrent kernel. 16 blocks per direction, each owns a
// 32-hidden slice; Wh slice (128 KB bf16) lives in LDS for all 512 steps.
// h tile is DMA'd into LDS by the TDM each step; device barrier per step.
// ---------------------------------------------------------------------------
__global__ void __launch_bounds__(TPB)
lstm_rec_kernel(const float* __restrict__ Wh_f, const float* __restrict__ Wh_b,
                const float* __restrict__ xg_f, const float* __restrict__ xg_b,
                const int* __restrict__ lengths,
                bf16* __restrict__ hbuf, float* __restrict__ out,
                unsigned* __restrict__ bar) {
  __shared__ bf16 sB[KS_ * NT_ * 32 * 16];   // 128 KB resident Wh slice
  __shared__ bf16 sA[32 * H_];               // 32 KB h tile; gate tile aliases it
  float* sG = (float*)sA;                    // 32x128 f32 gate tile (16 KB)
  const int slice = blockIdx.x;              // 0..15 -> hidden cols [c0, c0+32)
  const int dir   = blockIdx.y;
  const float* Wh = dir ? Wh_b : Wh_f;
  const float* XG = dir ? xg_b : xg_f;
  bf16* Hbuf = hbuf + (size_t)dir * 2 * B_ * H_;
  unsigned* cnt = bar + dir * 16;            // separate cacheline per direction
  const int c0 = slice * 32;
  load_B_slice(Wh, G4_, make_int4(c0, H_ + c0, 2 * H_ + c0, 3 * H_ + c0), sB);
  const int tid  = threadIdx.x;
  const int lane = tid & 31, wave = tid >> 5, half = lane >> 4;
  const int m_ew = tid >> 3;                 // batch row for elementwise (0..31)
  const int hl0  = (tid & 7) * 4;            // 4 hidden units per thread
  const int L    = lengths[m_ew];
  float c[4] = {0.f, 0.f, 0.f, 0.f};         // cell state lives in registers
  __syncthreads();
  for (int s = 0; s < T_; ++s) {
    const int par = s & 1;
    if (wave == 0) {                         // TDM: DMA full h (32 KB) into LDS
      tdm_load_b16_1d(Hbuf + (size_t)par * B_ * H_, sA, B_ * H_);
      __builtin_amdgcn_s_wait_tensorcnt(0);
    }
    __syncthreads();
    v8f acc0 = {}, acc1 = {};
    compute_tile(sA, sB, acc0, acc1);        // gates_pre = h @ Wh[:, slice]
    __syncthreads();                         // all waves done reading sA
    const int ncol = wave * 16 + (lane & 15);
    #pragma unroll
    for (int r = 0; r < 8; ++r) {
      sG[(r + 8 * half) * 128 + ncol]      = acc0[r];
      sG[(16 + r + 8 * half) * 128 + ncol] = acc1[r];
    }
    __syncthreads();
    const float* xgrow = XG + ((size_t)m_ew * T_ + s) * G4_;
    if (s + 1 < T_)                          // global_prefetch of next gate row
      __builtin_prefetch(XG + ((size_t)m_ew * T_ + (s + 1)) * G4_ + c0, 0, 0);
    bf16* hdst = Hbuf + (size_t)(par ^ 1) * B_ * H_;
    const int tout = dir ? ((L + T_ - 1 - s) & (T_ - 1)) : s;   // un-flip bwd
    float* orow = out + ((size_t)m_ew * T_ + tout) * (2 * H_) + dir * H_ + c0;
    #pragma unroll
    for (int q = 0; q < 4; ++q) {
      const int hl = hl0 + q;
      const float gi = sG[m_ew * 128 + hl]      + xgrow[0 * H_ + c0 + hl];
      const float gf = sG[m_ew * 128 + 32 + hl] + xgrow[1 * H_ + c0 + hl];
      const float gg = sG[m_ew * 128 + 64 + hl] + xgrow[2 * H_ + c0 + hl];
      const float go = sG[m_ew * 128 + 96 + hl] + xgrow[3 * H_ + c0 + hl];
      const float iv = 1.f / (1.f + __expf(-gi));
      const float fv = 1.f / (1.f + __expf(-gf));
      const float gv = tanhf(gg);
      const float ov = 1.f / (1.f + __expf(-go));
      c[q] = fv * c[q] + iv * gv;
      const float hv = ov * tanhf(c[q]);
      orow[hl] = hv;
      hdst[(size_t)m_ew * H_ + c0 + hl] = (bf16)hv;
    }
    __threadfence();
    __syncthreads();
    if (tid == 0) {                          // 16-block device barrier per dir
      __hip_atomic_fetch_add(cnt, 1u, __ATOMIC_RELEASE, __HIP_MEMORY_SCOPE_AGENT);
      const unsigned tgt = (unsigned)(s + 1) * (unsigned)NBLK_PER_DIR;
      while (__hip_atomic_load(cnt, __ATOMIC_ACQUIRE, __HIP_MEMORY_SCOPE_AGENT) < tgt) {
        __builtin_amdgcn_s_sleep(1);
      }
    }
    __syncthreads();
  }
}

// ---------------------------------------------------------------------------
extern "C" void kernel_launch(void* const* d_in, const int* in_sizes, int n_in,
                              void* d_out, int out_size, void* d_ws, size_t ws_size,
                              hipStream_t stream) {
  const float* x       = (const float*)d_in[0];
  const int*   lengths = (const int*)d_in[1];
  const float* Wx_f    = (const float*)d_in[2];
  const float* Wh_f    = (const float*)d_in[3];
  const float* b_f     = (const float*)d_in[4];
  const float* Wx_b    = (const float*)d_in[5];
  const float* Wh_b    = (const float*)d_in[6];
  const float* b_b     = (const float*)d_in[7];
  float* out = (float*)d_out;

  char* ws = (char*)d_ws;
  size_t off = 0;
  bf16*  xb   = (bf16*)(ws + off);  off += (size_t)B_ * T_ * D_ * 2;
  bf16*  xrb  = (bf16*)(ws + off);  off += (size_t)B_ * T_ * D_ * 2;
  float* xg_f = (float*)(ws + off); off += (size_t)B_ * T_ * G4_ * 4;
  float* xg_b = (float*)(ws + off); off += (size_t)B_ * T_ * G4_ * 4;
  bf16*  hbuf = (bf16*)(ws + off);  off += (size_t)2 * 2 * B_ * H_ * 2;
  unsigned* bar = (unsigned*)(ws + off); off += 256;

  prep_kernel<<<(B_ * T_ * D_) / TPB, TPB, 0, stream>>>(x, lengths, xb, xrb, hbuf, bar);

  dim3 g1(16, 64, 2);   // n-slices x m-chunks x directions
  proj_gemm_kernel<<<g1, TPB, 0, stream>>>(xb, xrb, Wx_f, Wx_b, b_f, b_b, xg_f, xg_b);

  dim3 g2(NBLK_PER_DIR, 2);
  lstm_rec_kernel<<<g2, TPB, 0, stream>>>(Wh_f, Wh_b, xg_f, xg_b, lengths, hbuf, out, bar);
}
